// molecular_conv_38826504356553
// MI455X (gfx1250) — matrix operand order repaired and checked by reading, not compiled, not run on hardware
//
#include <hip/hip_runtime.h>

#define NN 50000
#define NE 500000
#define NG 2000
#define HD 128
#define IND 64
#define EDD 16

typedef __attribute__((ext_vector_type(16))) _Float16 v16h;
typedef __attribute__((ext_vector_type(8)))  _Float16 v8h;
typedef __attribute__((ext_vector_type(8)))  float    v8f;
typedef __attribute__((ext_vector_type(4)))  unsigned u32x4;
typedef __attribute__((ext_vector_type(8)))  int      i32x8;
typedef __attribute__((ext_vector_type(4)))  int      i32x4;

// ---- Tensor Data Mover availability (device pass only; arity per toolchain)
#if defined(__HIP_DEVICE_COMPILE__) && defined(__has_builtin)
#if __has_builtin(__builtin_amdgcn_tensor_load_to_lds)
#define HAVE_TDM 1
#endif
#endif
#if defined(__has_include)
#if __has_include(<hip/amd_detail/amd_gfx1250_TDM.h>)
#define TDM_ARITY 6
#else
#define TDM_ARITY 5
#endif
#else
#define TDM_ARITY 5
#endif

// ---------------------------------------------------------------- WMMA helpers

__device__ __forceinline__ v8f wmma32(v16h a, v16h b, v8f c) {
  // v_wmma_f32_16x16x32_f16 : D = A(16x32 f16) * B(32x16 f16) + C(16x16 f32)
  return __builtin_amdgcn_wmma_f32_16x16x32_f16(false, a, false, b, (short)0, c,
                                                false, false);
}

// A fragment: lane(half = lane>>4, m = lane&15) holds row m:
//   a[0..7]  = A[m][k0 + 8*half .. +7]
//   a[8..15] = A[m][k0 + 16 + 8*half .. +7]
__device__ __forceinline__ v16h load_afrag(const _Float16* rowbase, int k0, int half) {
  const v8h lo = *reinterpret_cast<const v8h*>(rowbase + k0 + 8 * half);
  const v8h hi = *reinterpret_cast<const v8h*>(rowbase + k0 + 16 + 8 * half);
  v16h a;
#pragma unroll
  for (int i = 0; i < 8; ++i) { a[i] = lo[i]; a[i + 8] = hi[i]; }
  return a;
}

// B fragment: B[k][n] = W[n][k] (weights stored [out][in], i.e. already B^T),
// lane(half, n) holds b[0..7] = W[n][k0 + 8*half .. +7], b[8..15] = W[n][k0 + 16 + 8*half .. +7].
__device__ __forceinline__ v16h load_bfrag(const _Float16* w, int ldw, int n,
                                           int k0, int half) {
  const _Float16* p = w + (size_t)n * ldw + k0 + 8 * half;
  const v8h lo = *reinterpret_cast<const v8h*>(p);
  const v8h hi = *reinterpret_cast<const v8h*>(p + 16);
  v16h b;
#pragma unroll
  for (int i = 0; i < 8; ++i) { b[i] = lo[i]; b[i + 8] = hi[i]; }
  return b;
}

__device__ __forceinline__ float sigm(float x) { return 1.0f / (1.0f + expf(-x)); }
__device__ __forceinline__ float leakyf(float x) { return x >= 0.0f ? x : 0.01f * x; }
__device__ __forceinline__ float eluf(float x) { return x > 0.0f ? x : expf(x) - 1.0f; }

// ---------------------------------------------------------------- conversions

__global__ void cvt_f32_f16(const float* __restrict__ in, _Float16* __restrict__ out, int n) {
  int i = blockIdx.x * 256 + threadIdx.x;
  if (i < n) out[i] = (_Float16)in[i];
}

// nd_lin1_w : [128][144] f32 -> [128][160] f16, zero-padded K tail
__global__ void pad_cvt_w1(const float* __restrict__ in, _Float16* __restrict__ out) {
  int i = blockIdx.x * 256 + threadIdx.x;
  if (i >= HD * 160) return;
  int r = i / 160, c = i % 160;
  out[i] = (c < HD + EDD) ? (_Float16)in[r * (HD + EDD) + c] : (_Float16)0.0f;
}

// ---------------------------------------------------------------- node GEMM
// out[N x Hout] = act(A_f16[N x K] @ W_f16[Hout x K]^T + bias)
// grid = rows/16, block = 32 * (Hout/16)
__global__ void node_gemm(const _Float16* __restrict__ A, int K,
                          const _Float16* __restrict__ W,
                          const float* __restrict__ bias,
                          float* __restrict__ out32, _Float16* __restrict__ out16,
                          int Hout, int act /*0 none, 1 leaky*/) {
  const int row0 = blockIdx.x * 16;
  const int wave = threadIdx.x >> 5, lane = threadIdx.x & 31;
  const int half = lane >> 4, n = lane & 15;
  const _Float16* arow = A + (size_t)(row0 + (lane & 15)) * K;
  v8f acc = {};
  for (int k0 = 0; k0 < K; k0 += 32)
    acc = wmma32(load_afrag(arow, k0, half), load_bfrag(W, K, wave * 16 + n, k0, half), acc);
  const int col = wave * 16 + n;
  const float bv = bias ? bias[col] : 0.0f;
#pragma unroll
  for (int r = 0; r < 8; ++r) {
    int row = row0 + r + 8 * half;
    float v = acc[r] + bv;
    if (act == 1) v = leakyf(v);
    out32[(size_t)row * Hout + col] = v;
    if (out16) out16[(size_t)row * Hout + col] = (_Float16)v;
  }
}

// ---------------------------------------------------------------- edge conv
// per block: 16 edges. m = leaky([x1[src]|eattr] @ W1p^T) @ W2^T ; atomic scatter to hsum[dst]
// The 16-row gather of x1f16 is offloaded to the Tensor Data Mover (gather mode,
// 16-bit row indices, pad_enable to produce the 320B LDS row stride).
__global__ void edge_conv(const int* __restrict__ src, const int* __restrict__ dst,
                          const _Float16* __restrict__ x1f16,
                          const float* __restrict__ eattr,
                          const _Float16* __restrict__ w1p /*128x160*/,
                          const _Float16* __restrict__ w2 /*128x128*/,
                          float* __restrict__ hsum) {
  __shared__ __align__(16) _Float16 A1[16][160];
  __shared__ __align__(16) _Float16 M1[16][HD];
  __shared__ int sDst[16];
  const int e0 = blockIdx.x * 16;
  const int t = threadIdx.x;
  if (t < 16) sDst[t] = dst[e0 + t];

#if defined(HAVE_TDM)
  if (t < 32) {  // wave 0 issues one TDM gather: 16 rows x 256B of x1f16 -> A1[.][0..127]
    const unsigned lds_base = (unsigned)(size_t)&A1[0][0];
    const unsigned long long ga = (unsigned long long)(size_t)x1f16;
    u32x4 g0;
    g0[0] = 0x80000001u;                                   // count=1, gather_mode=1, 16b idx
    g0[1] = lds_base;                                      // lds_addr (bytes)
    g0[2] = (unsigned)ga;                                  // global_addr[31:0]
    g0[3] = (unsigned)((ga >> 32) & 0x01FFFFFFu) | 0x80000000u;  // addr[56:32] | type=2
    i32x8 g1;
    g1[0] = (int)((2u << 16) | (1u << 20) | (5u << 22) | (15u << 25));
    //        data_size=4B    pad_enable    pad_int=64dw   pad_amt=16dw
    g1[1] = (int)(64u << 16);            // tensor_dim0 = 64 (4B units) low16 @bit48
    g1[2] = (int)((unsigned)NN << 16);   // tensor_dim0 hi=0 | tensor_dim1 low16 @bit80
    g1[3] = (int)((NN >> 16) | (64u << 16)); // tensor_dim1 hi | tile_dim0 = 64 units
    g1[4] = 16;                          // tile_dim1 = #valid indices
    g1[5] = 64;                          // tensor_dim0_stride = 64 (4B units)
    g1[6] = 0;
    g1[7] = 0;
    int idx[16];
#pragma unroll
    for (int k = 0; k < 16; ++k) idx[k] = src[e0 + k];     // block-uniform -> s_load
    i32x4 g2, g3;
#pragma unroll
    for (int k = 0; k < 4; ++k) {
      g2[k] = (idx[2 * k] & 0xFFFF) | (idx[2 * k + 1] << 16);
      g3[k] = (idx[8 + 2 * k] & 0xFFFF) | (idx[8 + 2 * k + 1] << 16);
    }
#if TDM_ARITY == 6
    i32x8 gz = {};
    __builtin_amdgcn_tensor_load_to_lds(g0, g1, g2, g3, gz, 0);
#else
    __builtin_amdgcn_tensor_load_to_lds(g0, g1, g2, g3, 0);
#endif
    __builtin_amdgcn_s_wait_tensorcnt(0);
  }
  // all threads fill the padded region: cols 128..143 = edge_attr, 144..159 = 0
  for (int i = t; i < 16 * 32; i += 256) {
    int r = i >> 5, c = HD + (i & 31);
    A1[r][c] = (c < HD + EDD) ? (_Float16)eattr[(size_t)(e0 + r) * EDD + (c - HD)]
                              : (_Float16)0.0f;
  }
#else
  __shared__ int sSrc[16];
  if (t < 16) sSrc[t] = src[e0 + t];
  __syncthreads();
  for (int i = t; i < 16 * 160; i += 256) {
    int r = i / 160, c = i % 160;
    _Float16 v;
    if (c < HD)            v = x1f16[(size_t)sSrc[r] * HD + c];
    else if (c < HD + EDD) v = (_Float16)eattr[(size_t)(e0 + r) * EDD + (c - HD)];
    else                   v = (_Float16)0.0f;
    A1[r][c] = v;
  }
#endif
  __syncthreads();

  const int wave = t >> 5, lane = t & 31, half = lane >> 4, n = lane & 15;
  const _Float16* arow = &A1[lane & 15][0];
  v8f acc = {};
  for (int k0 = 0; k0 < 160; k0 += 32)
    acc = wmma32(load_afrag(arow, k0, half), load_bfrag(w1p, 160, wave * 16 + n, k0, half), acc);
#pragma unroll
  for (int r = 0; r < 8; ++r)
    M1[r + 8 * half][wave * 16 + n] = (_Float16)leakyf(acc[r]);
  __syncthreads();
  const _Float16* arow2 = &M1[lane & 15][0];
  v8f acc2 = {};
  for (int k0 = 0; k0 < HD; k0 += 32)
    acc2 = wmma32(load_afrag(arow2, k0, half), load_bfrag(w2, HD, wave * 16 + n, k0, half), acc2);
#pragma unroll
  for (int r = 0; r < 8; ++r)
    atomicAdd(&hsum[(size_t)sDst[r + 8 * half] * HD + wave * 16 + n], acc2[r]);
}

// ---------------------------------------------------------------- elu + bias (conv epilogue)
__global__ void elu_bias(const float* __restrict__ hsum, const float* __restrict__ bias,
                         _Float16* __restrict__ h16) {
  int i = blockIdx.x * 256 + threadIdx.x;
  if (i >= NN * HD) return;
  h16[i] = (_Float16)eluf(hsum[i] + bias[i & (HD - 1)]);
}

// ---------------------------------------------------------------- fused GRU cell + ReLU
// xcur = relu(gru_cell(x = xh16, h = (hid16, hid32)))
__global__ void gru_fused(const _Float16* __restrict__ xh16,
                          const _Float16* hid16, const float* hid32,
                          const _Float16* __restrict__ wih, const _Float16* __restrict__ whh,
                          const float* __restrict__ bih, const float* __restrict__ bhh,
                          float* out32, _Float16* out16) {
  __shared__ float gi[16][3 * HD];
  __shared__ float gh[16][3 * HD];
  const int row0 = blockIdx.x * 16;
  const int wave = threadIdx.x >> 5, lane = threadIdx.x & 31;
  const int half = lane >> 4, n = lane & 15;
  const _Float16* ax = xh16 + (size_t)(row0 + (lane & 15)) * HD;
  const _Float16* ah = hid16 + (size_t)(row0 + (lane & 15)) * HD;
  for (int tt = 0; tt < 3; ++tt) {
    const int colt = wave * 3 + tt, col = colt * 16 + n;
    v8f acc = {};
    for (int k0 = 0; k0 < HD; k0 += 32)
      acc = wmma32(load_afrag(ax, k0, half), load_bfrag(wih, HD, col, k0, half), acc);
#pragma unroll
    for (int r = 0; r < 8; ++r) gi[r + 8 * half][col] = acc[r] + bih[col];
    v8f acc2 = {};
    for (int k0 = 0; k0 < HD; k0 += 32)
      acc2 = wmma32(load_afrag(ah, k0, half), load_bfrag(whh, HD, col, k0, half), acc2);
#pragma unroll
    for (int r = 0; r < 8; ++r) gh[r + 8 * half][col] = acc2[r] + bhh[col];
  }
  __syncthreads();
  for (int i = threadIdx.x; i < 16 * HD; i += 256) {
    int r = i >> 7, c = i & (HD - 1);
    float rg = sigm(gi[r][c] + gh[r][c]);
    float z  = sigm(gi[r][c + HD] + gh[r][c + HD]);
    float ng = tanhf(gi[r][c + 2 * HD] + rg * gh[r][c + 2 * HD]);
    float hp = hid32[(size_t)(row0 + r) * HD + c];
    float v  = fmaxf((1.0f - z) * ng + z * hp, 0.0f);
    out32[(size_t)(row0 + r) * HD + c] = v;
    out16[(size_t)(row0 + r) * HD + c] = (_Float16)v;
  }
}

// ---------------------------------------------------------------- GAT attention pieces

__global__ void attn_node_scores(const float* __restrict__ xp,
                                 const float* __restrict__ asrc, const float* __restrict__ adst,
                                 float* __restrict__ ssrc, float* __restrict__ sdst) {
  int nid = blockIdx.x * 256 + threadIdx.x;
  if (nid >= NN) return;
  const float* r = xp + (size_t)nid * HD;
  float s1 = 0.0f, s2 = 0.0f;
  for (int c = 0; c < HD; ++c) { float v = r[c]; s1 += v * asrc[c]; s2 += v * adst[c]; }
  ssrc[nid] = s1; sdst[nid] = s2;
}

__global__ void edge_attn_max(const int* __restrict__ src, const int* __restrict__ dst,
                              const float* __restrict__ ssrc, const float* __restrict__ sdst,
                              float* __restrict__ a_e, unsigned* __restrict__ amax_ord) {
  int e = blockIdx.x * 256 + threadIdx.x;
  if (e >= NE) return;
  float a = leakyf(ssrc[src[e]] + sdst[dst[e]]);
  a_e[e] = a;
  unsigned bits = __float_as_uint(a);
  unsigned ord = (bits & 0x80000000u) ? ~bits : (bits | 0x80000000u);
  atomicMax(&amax_ord[dst[e]], ord);
}

__global__ void edge_attn_exp(const int* __restrict__ dst, const float* __restrict__ a_e,
                              const unsigned* __restrict__ amax_ord,
                              float* __restrict__ ae, float* __restrict__ denom) {
  int e = blockIdx.x * 256 + threadIdx.x;
  if (e >= NE) return;
  unsigned o = amax_ord[dst[e]];
  unsigned bits = (o & 0x80000000u) ? (o & 0x7fffffffu) : ~o;
  float v = expf(a_e[e] - __uint_as_float(bits));
  ae[e] = v;
  atomicAdd(&denom[dst[e]], v);
}

__global__ void edge_attn_scatter(const int* __restrict__ src, const int* __restrict__ dst,
                                  const float* __restrict__ ae, const float* __restrict__ denom,
                                  const float* __restrict__ xp, float* __restrict__ hsum) {
  int idx = blockIdx.x * 256 + threadIdx.x;  // NE * 32 threads, 4 cols each
  int e = idx >> 5;
  if (e >= NE) return;
  int c = (idx & 31) << 2;
  int d = dst[e];
  float coef = ae[e] / denom[d];
  const float4 v = *reinterpret_cast<const float4*>(xp + (size_t)src[e] * HD + c);
  float* b = hsum + (size_t)d * HD + c;
  atomicAdd(b + 0, coef * v.x); atomicAdd(b + 1, coef * v.y);
  atomicAdd(b + 2, coef * v.z); atomicAdd(b + 3, coef * v.w);
}

// ---------------------------------------------------------------- readout

__global__ void batch_scatter(const float* __restrict__ xcur, const int* __restrict__ batch,
                              float* __restrict__ agg) {
  int idx = blockIdx.x * 256 + threadIdx.x;  // NN * 32
  int nid = idx >> 5;
  if (nid >= NN) return;
  int c = (idx & 31) << 2;
  const float4 v = *reinterpret_cast<const float4*>(xcur + (size_t)nid * HD + c);
  float* b = agg + (size_t)batch[nid] * HD + c;
  atomicAdd(b + 0, v.x); atomicAdd(b + 1, v.y); atomicAdd(b + 2, v.z); atomicAdd(b + 3, v.w);
}

__global__ void relu_copy(const float* __restrict__ in, float* __restrict__ out, int n) {
  int i = blockIdx.x * 256 + threadIdx.x;
  if (i < n) out[i] = fmaxf(in[i], 0.0f);
}

// one GIN + LSTM timestep on 16-graph tiles, in-place update of `out`
__global__ void lstm_fused(const float* __restrict__ agg, float* out,
                           const _Float16* __restrict__ ginw, const float* __restrict__ ginb,
                           const _Float16* __restrict__ wih, const _Float16* __restrict__ whh,
                           const float* __restrict__ bih, const float* __restrict__ bhh) {
  __shared__ __align__(16) _Float16 s16[16][HD];   // f16(out + agg)
  __shared__ __align__(16) _Float16 o16[16][HD];   // f16(out)
  __shared__ __align__(16) _Float16 h16[16][HD];   // f16(h) after GIN
  __shared__ float hph[16][HD];                    // h f32 (initial cell state)
  __shared__ float g[16][4 * HD];                  // LSTM gate pre-activations
  const int row0 = blockIdx.x * 16;
  const int t = threadIdx.x;
  for (int i = t; i < 16 * HD; i += 256) {
    int r = i >> 7, c = i & (HD - 1);
    float o = out[(size_t)(row0 + r) * HD + c];
    o16[r][c] = (_Float16)o;
    s16[r][c] = (_Float16)(o + agg[(size_t)(row0 + r) * HD + c]);
  }
  __syncthreads();
  const int wave = t >> 5, lane = t & 31, half = lane >> 4, n = lane & 15;
  // GIN: h = elu((out+agg) @ ginw^T + ginb)
  {
    const _Float16* arow = &s16[lane & 15][0];
    v8f acc = {};
    for (int k0 = 0; k0 < HD; k0 += 32)
      acc = wmma32(load_afrag(arow, k0, half), load_bfrag(ginw, HD, wave * 16 + n, k0, half), acc);
    const int col = wave * 16 + n;
#pragma unroll
    for (int r = 0; r < 8; ++r) {
      float v = eluf(acc[r] + ginb[col]);
      hph[r + 8 * half][col] = v;
      h16[r + 8 * half][col] = (_Float16)v;
    }
  }
  __syncthreads();
  // LSTM gates: g = out @ wih^T + bih + h @ whh^T + bhh   (4H = 512 cols)
  const _Float16* ao = &o16[lane & 15][0];
  const _Float16* ah = &h16[lane & 15][0];
  for (int j = 0; j < 4; ++j) {
    const int col = (wave + 8 * j) * 16 + n;
    v8f acc = {};
    for (int k0 = 0; k0 < HD; k0 += 32)
      acc = wmma32(load_afrag(ao, k0, half), load_bfrag(wih, HD, col, k0, half), acc);
    for (int k0 = 0; k0 < HD; k0 += 32)
      acc = wmma32(load_afrag(ah, k0, half), load_bfrag(whh, HD, col, k0, half), acc);
#pragma unroll
    for (int r = 0; r < 8; ++r) g[r + 8 * half][col] = acc[r] + bih[col] + bhh[col];
  }
  __syncthreads();
  for (int i = t; i < 16 * HD; i += 256) {
    int r = i >> 7, c = i & (HD - 1);
    float ig = g[r][c], fg = g[r][c + HD], gg = g[r][c + 2 * HD], og = g[r][c + 3 * HD];
    float cc = hph[r][c];
    float c2 = sigm(fg) * cc + sigm(ig) * tanhf(gg);
    out[(size_t)(row0 + r) * HD + c] = sigm(og) * tanhf(c2);
  }
}

__global__ void final_proj(const float* __restrict__ out, const float* __restrict__ w,
                           const float* __restrict__ b, float* __restrict__ y) {
  int gid = blockIdx.x * 256 + threadIdx.x;
  if (gid >= NG) return;
  const float* r = out + (size_t)gid * HD;
  float s = b[0];
  for (int c = 0; c < HD; ++c) s += r[c] * w[c];
  y[gid] = s;
}

// ---------------------------------------------------------------- launcher

extern "C" void kernel_launch(void* const* d_in, const int* in_sizes, int n_in,
                              void* d_out, int out_size, void* d_ws, size_t ws_size,
                              hipStream_t stream) {
  (void)in_sizes; (void)n_in; (void)out_size;
  const float* x       = (const float*)d_in[0];
  const int*   ei      = (const int*)d_in[1];
  const float* eattr   = (const float*)d_in[2];
  const int*   batch   = (const int*)d_in[3];
  const float* lin1_w  = (const float*)d_in[4];
  const float* lin1_b  = (const float*)d_in[5];
  const float* nd1_w   = (const float*)d_in[6];
  const float* nd2_w   = (const float*)d_in[7];
  const float* nd_bias = (const float*)d_in[8];
  const float* g0_wih  = (const float*)d_in[9];
  const float* g0_whh  = (const float*)d_in[10];
  const float* g0_bih  = (const float*)d_in[11];
  const float* g0_bhh  = (const float*)d_in[12];
  const float* gat_w   = (const float*)d_in[13];
  const float* gat_as  = (const float*)d_in[14];
  const float* gat_ad  = (const float*)d_in[15];
  const float* gat_b   = (const float*)d_in[16];
  const float* gl_wih  = (const float*)d_in[17];
  const float* gl_whh  = (const float*)d_in[18];
  const float* gl_bih  = (const float*)d_in[19];
  const float* gl_bhh  = (const float*)d_in[20];
  const float* gin_w   = (const float*)d_in[21];
  const float* gin_b   = (const float*)d_in[22];
  const float* ls_wih  = (const float*)d_in[23];
  const float* ls_whh  = (const float*)d_in[24];
  const float* ls_bih  = (const float*)d_in[25];
  const float* ls_bhh  = (const float*)d_in[26];
  const float* lin2_w  = (const float*)d_in[27];
  const float* lin2_b  = (const float*)d_in[28];
  float* y = (float*)d_out;

  const int* esrc = ei;
  const int* edst = ei + NE;

  // ---- workspace bump allocator
  char* base = (char*)d_ws;
  size_t off = 0;
  auto alloc = [&](size_t bytes) -> void* {
    void* r = base + off;
    off = (off + bytes + 255) & ~(size_t)255;
    return r;
  };
  _Float16* xf16   = (_Float16*)alloc((size_t)NN * IND * 2);
  float*    x1_32  = (float*)   alloc((size_t)NN * HD * 4);  // reused as xp_32 after GRU0
  _Float16* x1_16  = (_Float16*)alloc((size_t)NN * HD * 2);
  float*    hsum   = (float*)   alloc((size_t)NN * HD * 4);
  _Float16* h16    = (_Float16*)alloc((size_t)NN * HD * 2);
  float*    xc32   = (float*)   alloc((size_t)NN * HD * 4);
  _Float16* xc16   = (_Float16*)alloc((size_t)NN * HD * 2);
  float*    ssrc   = (float*)   alloc((size_t)NN * 4);
  float*    sdst   = (float*)   alloc((size_t)NN * 4);
  unsigned* amax   = (unsigned*)alloc((size_t)NN * 4);
  float*    denom  = (float*)   alloc((size_t)NN * 4);
  float*    a_e    = (float*)   alloc((size_t)NE * 4);
  float*    ae     = (float*)   alloc((size_t)NE * 4);
  float*    agg    = (float*)   alloc((size_t)NG * HD * 4);
  float*    out32  = (float*)   alloc((size_t)NG * HD * 4);
  _Float16* lin1h  = (_Float16*)alloc((size_t)HD * IND * 2);
  _Float16* w1ph   = (_Float16*)alloc((size_t)HD * 160 * 2);
  _Float16* w2h    = (_Float16*)alloc((size_t)HD * HD * 2);
  _Float16* g0wihh = (_Float16*)alloc((size_t)3 * HD * HD * 2);
  _Float16* g0whhh = (_Float16*)alloc((size_t)3 * HD * HD * 2);
  _Float16* gatwh  = (_Float16*)alloc((size_t)2 * HD * HD * 2);
  _Float16* glwihh = (_Float16*)alloc((size_t)2 * 3 * HD * HD * 2);
  _Float16* glwhhh = (_Float16*)alloc((size_t)2 * 3 * HD * HD * 2);
  _Float16* ginwh  = (_Float16*)alloc((size_t)HD * HD * 2);
  _Float16* lswihh = (_Float16*)alloc((size_t)4 * HD * HD * 2);
  _Float16* lswhhh = (_Float16*)alloc((size_t)4 * HD * HD * 2);
  if (off > ws_size) return;

  auto cvt = [&](const float* s, _Float16* d, int n) {
    cvt_f32_f16<<<dim3((n + 255) / 256), dim3(256), 0, stream>>>(s, d, n);
  };

  // ---- weight / input conversions
  cvt(x, xf16, NN * IND);
  cvt(lin1_w, lin1h, HD * IND);
  pad_cvt_w1<<<dim3((HD * 160 + 255) / 256), dim3(256), 0, stream>>>(nd1_w, w1ph);
  cvt(nd2_w, w2h, HD * HD);
  cvt(g0_wih, g0wihh, 3 * HD * HD);
  cvt(g0_whh, g0whhh, 3 * HD * HD);
  cvt(gat_w, gatwh, 2 * HD * HD);
  cvt(gl_wih, glwihh, 2 * 3 * HD * HD);
  cvt(gl_whh, glwhhh, 2 * 3 * HD * HD);
  cvt(gin_w, ginwh, HD * HD);
  cvt(ls_wih, lswihh, 4 * HD * HD);
  cvt(ls_whh, lswhhh, 4 * HD * HD);

  // ---- x1 = leaky(x @ lin1^T + b)
  node_gemm<<<dim3(NN / 16), dim3(256), 0, stream>>>(xf16, IND, lin1h, lin1_b,
                                                     x1_32, x1_16, HD, 1);
  // ---- nd_conv: edge MLP + scatter-add, then elu+bias
  hipMemsetAsync(hsum, 0, (size_t)NN * HD * 4, stream);
  edge_conv<<<dim3(NE / 16), dim3(256), 0, stream>>>(esrc, edst, x1_16, eattr,
                                                     w1ph, w2h, hsum);
  elu_bias<<<dim3(NN * HD / 256), dim3(256), 0, stream>>>(hsum, nd_bias, h16);
  // ---- GRU0: xcur = relu(gru(h, x1))
  gru_fused<<<dim3(NN / 16), dim3(256), 0, stream>>>(h16, x1_16, x1_32,
                                                     g0wihh, g0whhh, g0_bih, g0_bhh,
                                                     xc32, xc16);
  // ---- GAT layers
  float* xp32 = x1_32;  // x1 dead after GRU0
  for (int l = 0; l < 2; ++l) {
    node_gemm<<<dim3(NN / 16), dim3(256), 0, stream>>>(xc16, HD, gatwh + (size_t)l * HD * HD,
                                                       (const float*)nullptr, xp32,
                                                       (_Float16*)nullptr, HD, 0);
    attn_node_scores<<<dim3((NN + 255) / 256), dim3(256), 0, stream>>>(
        xp32, gat_as + l * HD, gat_ad + l * HD, ssrc, sdst);
    hipMemsetAsync(amax, 0, (size_t)NN * 4, stream);
    edge_attn_max<<<dim3((NE + 255) / 256), dim3(256), 0, stream>>>(esrc, edst, ssrc, sdst,
                                                                    a_e, amax);
    hipMemsetAsync(denom, 0, (size_t)NN * 4, stream);
    edge_attn_exp<<<dim3((NE + 255) / 256), dim3(256), 0, stream>>>(edst, a_e, amax, ae, denom);
    hipMemsetAsync(hsum, 0, (size_t)NN * HD * 4, stream);
    edge_attn_scatter<<<dim3(NE * 32 / 256), dim3(256), 0, stream>>>(esrc, edst, ae, denom,
                                                                     xp32, hsum);
    elu_bias<<<dim3(NN * HD / 256), dim3(256), 0, stream>>>(hsum, gat_b + l * HD, h16);
    gru_fused<<<dim3(NN / 16), dim3(256), 0, stream>>>(h16, xc16, xc32,
                                                       glwihh + (size_t)l * 3 * HD * HD,
                                                       glwhhh + (size_t)l * 3 * HD * HD,
                                                       gl_bih + l * 3 * HD, gl_bhh + l * 3 * HD,
                                                       xc32, xc16);  // in-place safe
  }
  // ---- readout
  hipMemsetAsync(agg, 0, (size_t)NG * HD * 4, stream);
  batch_scatter<<<dim3(NN * 32 / 256), dim3(256), 0, stream>>>(xc32, batch, agg);
  relu_copy<<<dim3(NG * HD / 256), dim3(256), 0, stream>>>(agg, out32, NG * HD);
  for (int tstep = 0; tstep < 2; ++tstep)
    lstm_fused<<<dim3(NG / 16), dim3(256), 0, stream>>>(agg, out32, ginwh, gin_b,
                                                        lswihh, lswhhh, ls_bih, ls_bhh);
  final_proj<<<dim3((NG + 255) / 256), dim3(256), 0, stream>>>(out32, lin2_w, lin2_b, y);
}
